// MultiHeadSelection_10385230921927
// MI455X (gfx1250) — matrix operand order repaired and checked by reading, not compiled, not run on hardware
//
#include <hip/hip_runtime.h>
#include <hip/hip_fp16.h>
#include <stdint.h>

typedef __attribute__((ext_vector_type(16))) _Float16 v16h;
typedef __attribute__((ext_vector_type(2)))  __fp16   pk2;   // cvt_pkrtz result type
typedef __attribute__((ext_vector_type(8)))  float    v8f;
typedef __attribute__((ext_vector_type(4)))  unsigned int v4u;
typedef __attribute__((ext_vector_type(8)))  int      v8i_;
typedef __attribute__((ext_vector_type(4)))  int      v4i_;

#define DIM_H 768
#define DIM_K 256
#define DIM_P 50
#define DIM_S 256
#define DIM_B 8
#define NROWS (DIM_B * DIM_S)   // 2048

// ---------------------------------------------------------------------------
// Fast tanh: prefer CDNA5 v_tanh_f32 (TRANS op), else branch-free exp2/rcp.
// Output feeds f16 WMMA fragments, so ~1e-3 accuracy is more than enough.
// ---------------------------------------------------------------------------
__device__ __forceinline__ float fast_tanh(float x) {
#if defined(__has_builtin) && __has_builtin(__builtin_amdgcn_tanhf)
    return __builtin_amdgcn_tanhf(x);
#elif defined(__has_builtin) && __has_builtin(__builtin_amdgcn_tanh_f32)
    return __builtin_amdgcn_tanh_f32(x);
#else
    // tanh(x) = (e^2x - 1) / (e^2x + 1), clamped so exp2 never overflows
    float xc = fminf(fmaxf(x, -10.0f), 10.0f);
    float t  = __builtin_amdgcn_exp2f(xc * 2.8853900817779268f); // 2*log2(e)
    return (t - 1.0f) * __builtin_amdgcn_rcpf(t + 1.0f);
#endif
}

// ---------------------------------------------------------------------------
// Kernel 1: left = x @ u_a ; right = x @ w_a   (f32, LDS-tiled, 64x64 tiles)
// ---------------------------------------------------------------------------
__global__ __launch_bounds__(256) void gemm_lr_kernel(
    const float* __restrict__ x,    // [2048][768]
    const float* __restrict__ u_a,  // [768][256]
    const float* __restrict__ w_a,  // [768][256]
    float* __restrict__ left,       // [2048][256]
    float* __restrict__ right)      // [2048][256]
{
    const int sel = blockIdx.z;
    const float* __restrict__ W = sel ? w_a : u_a;
    float* __restrict__ O = sel ? right : left;

    const int k0 = blockIdx.x * 64;
    const int i0 = blockIdx.y * 64;

    __shared__ float xs[64][17];   // [i][h], padded
    __shared__ float ws[16][64];   // [h][k]

    const int t  = threadIdx.x;
    const int tx = t & 15;         // k micro index
    const int ty = t >> 4;         // i micro index

    float acc[4][4] = {};

    for (int h0 = 0; h0 < DIM_H; h0 += 16) {
        {
            const int rr = t >> 2;
            const int hh = (t & 3) * 4;
            float4 xv = *(const float4*)(x + (size_t)(i0 + rr) * DIM_H + h0 + hh);
            xs[rr][hh + 0] = xv.x; xs[rr][hh + 1] = xv.y;
            xs[rr][hh + 2] = xv.z; xs[rr][hh + 3] = xv.w;
        }
        {
            const int hr = t >> 4;
            const int cc = (t & 15) * 4;
            float4 wv = *(const float4*)(W + (size_t)(h0 + hr) * DIM_K + k0 + cc);
            ws[hr][cc + 0] = wv.x; ws[hr][cc + 1] = wv.y;
            ws[hr][cc + 2] = wv.z; ws[hr][cc + 3] = wv.w;
        }
        __syncthreads();

        #pragma unroll
        for (int h = 0; h < 16; ++h) {
            float xr[4], wr[4];
            #pragma unroll
            for (int a = 0; a < 4; ++a) xr[a] = xs[ty * 4 + a][h];
            #pragma unroll
            for (int c = 0; c < 4; ++c) wr[c] = ws[h][tx * 4 + c];
            #pragma unroll
            for (int a = 0; a < 4; ++a)
                #pragma unroll
                for (int c = 0; c < 4; ++c)
                    acc[a][c] = fmaf(xr[a], wr[c], acc[a][c]);
        }
        __syncthreads();
    }

    #pragma unroll
    for (int a = 0; a < 4; ++a)
        #pragma unroll
        for (int c = 0; c < 4; ++c)
            O[(size_t)(i0 + ty * 4 + a) * DIM_K + k0 + tx * 4 + c] = acc[a][c];
}

// ---------------------------------------------------------------------------
// Kernel 2: pack v (K x P, P padded to 64) into f16 WMMA B-fragment layout.
// B 32x16 f16 (ISA 7.12.2/7.12.5): lane L -> col n=L%16; VGPR v holds
// K = 2v,2v+1 (+8 if v>=4) (+8 if L>=16). Stored vfrag[(f*32+L)*8 + v], low half = even K.
// ---------------------------------------------------------------------------
__global__ __launch_bounds__(256) void vpack_kernel(
    const float* __restrict__ vsrc,   // [256][50]
    uint32_t* __restrict__ vfrag)     // [8192]
{
    const int d = blockIdx.x * 256 + threadIdx.x;   // 0..8191
    const int vv = d & 7;
    const int L  = (d >> 3) & 31;
    const int f  = d >> 8;
    const int kstep = f >> 2;
    const int nt    = f & 3;
    const int n  = nt * 16 + (L & 15);
    const int kl = 2 * vv + ((vv >= 4) ? 8 : 0) + ((L >> 4) ? 8 : 0);
    const int k  = kstep * 32 + kl;
    const float a = (n < DIM_P) ? vsrc[(size_t)k * DIM_P + n] : 0.0f;
    const float b = (n < DIM_P) ? vsrc[(size_t)(k + 1) * DIM_P + n] : 0.0f;
    union { pk2 h; uint32_t u; } pk;
    pk.h = __builtin_amdgcn_cvt_pkrtz(a, b);
    vfrag[d] = pk.u;
}

// ---------------------------------------------------------------------------
// Kernel 3: fused  scores[b,i,j,p] = tanh(left[b,i,k]+right[b,j,k]+b_s[k]) @ v
// Block: 256 thr = 8 waves; handles (b, 8 i's, one 16-wide j-tile).
// sR (right j-tile) staged by the Tensor Data Mover with LDS padding
// (4 dwords per 256) to reproduce the bank-spread stride-260 layout.
// ---------------------------------------------------------------------------
__global__ __launch_bounds__(256) void bilinear_tanh_wmma_kernel(
    const float* __restrict__ left,     // [2048][256]
    const float* __restrict__ right,    // [2048][256]
    const float* __restrict__ bs,       // [256]
    const uint32_t* __restrict__ vfrag, // [8192]
    float* __restrict__ out)            // [8][256][256][50]
{
    __shared__ float    sS[8][256];     // left[i]+b_s rows (broadcast reads)
    __shared__ float    sR[16][260];    // right j-tile, padded stride
    __shared__ uint32_t sV[8192];       // packed v fragments

    const int jt   = blockIdx.x;        // 0..15
    const int ib   = blockIdx.y;        // 0..31
    const int b    = blockIdx.z;        // 0..7
    const int t    = threadIdx.x;
    const int w    = t >> 5;            // wave id = local i
    const int lane = t & 31;

#if defined(__has_builtin) && __has_builtin(__builtin_amdgcn_tensor_load_to_lds)
    // --- stage sR via TDM (wave 0 issues; EXEC ignored for tensor ops) ---
    if (t < 32) {
        const uint64_t gaddr =
            (uint64_t)(uintptr_t)(right + (size_t)(b * DIM_S + jt * 16) * DIM_K);
        const uint32_t laddr = (uint32_t)(uintptr_t)&sR[0][0];
        v4u g0;
        g0[0] = 1u;                                   // count=1, user mode
        g0[1] = laddr;                                // lds_addr
        g0[2] = (uint32_t)gaddr;                      // global_addr[31:0]
        g0[3] = (uint32_t)((gaddr >> 32) & 0x1FFFFFFu) | (2u << 30); // type=2
        v8i_ g1;
        g1[0] = (int)((2u << 16)        // data_size = 4B
                    | (1u << 20)        // pad_enable
                    | (7u << 22)        // pad_interval: every 256 dwords
                    | (3u << 25));      // pad_amount: 4 dwords
        g1[1] = (int)(256u << 16);      // tensor_dim0 = 256 (K)
        g1[2] = (int)(2048u << 16);     // tensor_dim1 = 2048 rows
        g1[3] = (int)(256u << 16);      // tile_dim0 = 256
        g1[4] = 16;                     // tile_dim1 = 16 rows
        g1[5] = 256;                    // tensor_dim0_stride = 256
        g1[6] = 0;
        g1[7] = 0;
        v4i_ z4 = {0, 0, 0, 0};
#if __clang_major__ >= 23
        v8i_ z8 = {0, 0, 0, 0, 0, 0, 0, 0};
        __builtin_amdgcn_tensor_load_to_lds(g0, g1, z4, z4, z8, 0);
#else
        __builtin_amdgcn_tensor_load_to_lds(g0, g1, z4, z4, 0);
#endif
    }
#endif

    // --- stage sS: per-wave row = left[b, ib*8+w, :] + b_s ---
    {
        const int i = ib * 8 + w;
        const float4* src = (const float4*)(left + (size_t)(b * DIM_S + i) * DIM_K);
        const float4* bsv = (const float4*)bs;
        float4* dst = (float4*)&sS[w][0];
        #pragma unroll
        for (int q = 0; q < 2; ++q) {
            const int f4 = q * 32 + lane;
            float4 xl = src[f4];
            float4 xb = bsv[f4];
            float4 r;
            r.x = xl.x + xb.x; r.y = xl.y + xb.y;
            r.z = xl.z + xb.z; r.w = xl.w + xb.w;
            dst[f4] = r;
        }
    }
#if !(defined(__has_builtin) && __has_builtin(__builtin_amdgcn_tensor_load_to_lds))
    // --- manual fallback: stage sR with vector loads ---
    {
        const int r = t >> 4;
        const int j = jt * 16 + r;
        const float4* src = (const float4*)(right + (size_t)(b * DIM_S + j) * DIM_K);
        #pragma unroll
        for (int q = 0; q < 4; ++q) {
            const int f4 = (t & 15) + q * 16;   // 0..63
            float4 xv = src[f4];
            const int k = f4 * 4;
            sR[r][k + 0] = xv.x; sR[r][k + 1] = xv.y;
            sR[r][k + 2] = xv.z; sR[r][k + 3] = xv.w;
        }
    }
#endif
    // --- stage sV ---
    {
        const uint4* src = (const uint4*)vfrag;
        uint4* dst = (uint4*)sV;
        #pragma unroll
        for (int q = 0; q < 8; ++q) {
            const int idx = q * 256 + t;
            dst[idx] = src[idx];
        }
    }
#if defined(__has_builtin) && __has_builtin(__builtin_amdgcn_tensor_load_to_lds)
    if (t < 32) __builtin_amdgcn_s_wait_tensorcnt(0);
#endif
    __syncthreads();

    const int jloc = lane & 15;
    const int hi   = lane >> 4;
    const float* __restrict__ sSrow = &sS[w][0];
    const float* __restrict__ sRrow = &sR[jloc][0];

    v8f acc0 = {}, acc1 = {}, acc2 = {}, acc3 = {};

    #pragma unroll
    for (int kstep = 0; kstep < 8; ++kstep) {
        // A fragment 16x32 f16 (ISA 7.12.2): lane row m=lane%16;
        // VGPR v: K = 2v (+8 if v>=4) (+8 if lane>=16), pair (k, k+1).
        union { v16h v; pk2 p[8]; } af;
        #pragma unroll
        for (int vv = 0; vv < 8; ++vv) {
            const int kl = 2 * vv + ((vv >= 4) ? 8 : 0) + (hi ? 8 : 0);
            const int k  = kstep * 32 + kl;
            const float2 sv = *(const float2*)(sSrow + k);
            const float2 rv = *(const float2*)(sRrow + k);
            af.p[vv] = __builtin_amdgcn_cvt_pkrtz(fast_tanh(sv.x + rv.x),
                                                  fast_tanh(sv.y + rv.y));
        }
        const v16h b0 = *(const v16h*)&sV[((kstep * 4 + 0) * 32 + lane) * 8];
        const v16h b1 = *(const v16h*)&sV[((kstep * 4 + 1) * 32 + lane) * 8];
        const v16h b2 = *(const v16h*)&sV[((kstep * 4 + 2) * 32 + lane) * 8];
        const v16h b3 = *(const v16h*)&sV[((kstep * 4 + 3) * 32 + lane) * 8];
        acc0 = __builtin_amdgcn_wmma_f32_16x16x32_f16(false, af.v, false, b0, (short)0, acc0, false, false);
        acc1 = __builtin_amdgcn_wmma_f32_16x16x32_f16(false, af.v, false, b1, (short)0, acc1, false, false);
        acc2 = __builtin_amdgcn_wmma_f32_16x16x32_f16(false, af.v, false, b2, (short)0, acc2, false, false);
        acc3 = __builtin_amdgcn_wmma_f32_16x16x32_f16(false, af.v, false, b3, (short)0, acc3, false, false);
    }

    // --- store C (ISA layout): VGPR r -> row r (+8 hi lanes), col = lane%16 ---
    const int i = ib * 8 + w;
    const size_t rowbase = ((size_t)(b * DIM_S + i) * DIM_S + jt * 16);
    #pragma unroll
    for (int r = 0; r < 8; ++r) {
        const int row = r + hi * 8;
        const size_t o = (rowbase + row) * DIM_P;
        const float vals[4] = { acc0[r], acc1[r], acc2[r], acc3[r] };
        #pragma unroll
        for (int nt = 0; nt < 4; ++nt) {
            const int col = nt * 16 + jloc;
            if (col < DIM_P) out[o + col] = vals[nt];
        }
    }
}

// ---------------------------------------------------------------------------
extern "C" void kernel_launch(void* const* d_in, const int* in_sizes, int n_in,
                              void* d_out, int out_size, void* d_ws, size_t ws_size,
                              hipStream_t stream)
{
    (void)in_sizes; (void)n_in; (void)out_size; (void)ws_size;
    const float* x   = (const float*)d_in[0];
    const float* u_a = (const float*)d_in[1];
    const float* w_a = (const float*)d_in[2];
    const float* b_s = (const float*)d_in[3];
    const float* v   = (const float*)d_in[4];
    float* out = (float*)d_out;

    float* left  = (float*)d_ws;                       // 2048*256 f32
    float* right = left + (size_t)NROWS * DIM_K;       // 2048*256 f32
    uint32_t* vfrag = (uint32_t*)(right + (size_t)NROWS * DIM_K); // 8192 u32

    gemm_lr_kernel<<<dim3(DIM_K / 64, NROWS / 64, 2), 256, 0, stream>>>(
        x, u_a, w_a, left, right);
    vpack_kernel<<<dim3(32), 256, 0, stream>>>(v, vfrag);
    bilinear_tanh_wmma_kernel<<<dim3(16, 32, DIM_B), 256, 0, stream>>>(
        left, right, b_s, vfrag, out);
}
